// _OuterProductMean_34995393528036
// MI455X (gfx1250) — matrix operand order repaired and checked by reading, not compiled, not run on hardware
//
#include <hip/hip_runtime.h>

typedef __attribute__((ext_vector_type(16))) __bf16 v16bf;
typedef __attribute__((ext_vector_type(8)))  float  v8f;

#define S_DIM   256
#define N_SEQ   256
#define C_DIM   32
#define CZ_DIM  128
#define LN_EPS  1e-5f

// bf16 outer tile in LDS, B-fragment order: [j_loc][c(32)][d(32)] with padded
// j-stride of 1032 bf16 (2064 B) -> conflict-free ds_load_b128 in phase 2.
#define JSTRIDE_H 1032
#define LDS_ELEMS (16 * JSTRIDE_H)          // 33,024 B static LDS

static __device__ __forceinline__ v8f wmma_bf16(v16bf a, v16bf b, v8f c) {
  return __builtin_amdgcn_wmma_f32_16x16x32_bf16(false, a, false, b, (short)0, c, false, false);
}

// ---------------------------------------------------------------------------
// Stage 1: LayerNorm + W_ab projection; emit a/b already in WMMA fragment order
//
// a_frag layout: [i][kb(8)][ct(2)] fragments of 512 bf16.
//   A operand (16x32 MxK, bf16), per ISA 7.12.2:
//     lanes 0-15  : M = lane,     K in {0..7, 16..23}
//     lanes 16-31 : M = lane-16,  K in {8..15, 24..31}
// b_frag layout: [j][kb(8)][dt(2)] fragments of 512 bf16.
//   B operand (32x16 KxN, bf16): lane half selects K half; element e -> k.
// ---------------------------------------------------------------------------
__global__ void __launch_bounds__(256)
ln_proj_kernel(const float* __restrict__ m_si,
               const float* __restrict__ gamma,
               const float* __restrict__ beta,
               const float* __restrict__ W_ab,
               __bf16* __restrict__ a_frag,
               __bf16* __restrict__ b_frag) {
  const int lane = threadIdx.x & 31;
  const int wave = threadIdx.x >> 5;
  const int row  = blockIdx.x * 8 + wave;   // flattened (s, n)
  const int s    = row >> 8;
  const int n    = row & 255;

  float m = m_si[(size_t)row * C_DIM + lane];

  float sum = m, sumsq = m * m;
  #pragma unroll
  for (int off = 16; off > 0; off >>= 1) {
    sum   += __shfl_xor(sum,   off, 32);
    sumsq += __shfl_xor(sumsq, off, 32);
  }
  float mu  = sum * (1.0f / C_DIM);
  float var = sumsq * (1.0f / C_DIM) - mu * mu;
  float x   = (m - mu) * rsqrtf(var + LN_EPS) * gamma[lane] + beta[lane];

  float acc_a = 0.f, acc_b = 0.f;
  #pragma unroll
  for (int c = 0; c < C_DIM; ++c) {
    float xc = __shfl(x, c, 32);
    acc_a += xc * W_ab[lane * C_DIM + c];
    acc_b += xc * W_ab[(lane + C_DIM) * C_DIM + c];
  }

  const int k  = n & 31;
  const int kb = n >> 5;

  { // scatter a into A-fragment order
    int m16    = lane & 15;
    int ct     = lane >> 4;
    int hi     = (k >> 3) & 1;
    int lane_f = m16 + 16 * hi;
    int e      = (k & 7) + ((k & 16) ? 8 : 0);
    a_frag[(((size_t)s * 8 + kb) * 2 + ct) * 512 + lane_f * 16 + e] = (__bf16)acc_a;
  }
  { // scatter b into B-fragment order
    int nn     = lane & 15;
    int dt     = lane >> 4;
    int lane_f = nn + 16 * ((k >> 3) & 1);
    int e      = (k & 7) + ((k & 16) ? 8 : 0);
    b_frag[(((size_t)s * 8 + kb) * 2 + dt) * 512 + lane_f * 16 + e] = (__bf16)acc_b;
  }
}

// ---------------------------------------------------------------------------
// Stage 1b: repack W_out (128 x 1024, f32) into bf16 A-fragments:
//   wfrag[ztile(8)][kb(32)] fragments of 512 bf16.
// ---------------------------------------------------------------------------
__global__ void __launch_bounds__(256)
wout_prep_kernel(const float* __restrict__ W_out, __bf16* __restrict__ wfrag) {
  int t      = blockIdx.x * blockDim.x + threadIdx.x;  // 0 .. 131071
  int e      = t & 15;
  int lane_f = (t >> 4) & 31;
  int frag   = t >> 9;                                 // ztile*32 + kb
  int kb     = frag & 31;
  int ztile  = frag >> 5;
  int m16    = lane_f & 15;
  int hi     = lane_f >> 4;
  int z      = ztile * 16 + m16;
  int kloc   = e + 16 * hi;
  wfrag[t]   = (__bf16)W_out[(size_t)z * 1024 + kb * 32 + kloc];
}

// ---------------------------------------------------------------------------
// Stage 2: WG = (one i, block of 16 j). 8 waves, 2 j per wave.
// Phase 1: outer_ij = A_i^T B_j (32x32, K=256): 8 independent WMMA chains/wave;
//          result converted ONCE to bf16 and stashed in B-fragment LDS order.
// Phase 2: z[:,j] = W_out @ vec(outer) / N + b_out  (M=128, N=16, K=1024),
//          B operand loaded raw (v16bf) from LDS; coalesced float4 stores.
// ---------------------------------------------------------------------------
__global__ void __launch_bounds__(256)
outer_proj_kernel(const __bf16* __restrict__ a_frag,
                  const __bf16* __restrict__ b_frag,
                  const __bf16* __restrict__ wfrag,
                  const float*  __restrict__ b_out,
                  float* __restrict__ out) {
  __shared__ __bf16 outer_lds[LDS_ELEMS];   // [j_loc][c][d] bf16, padded j-stride

  const int lane = threadIdx.x & 31;
  const int wave = threadIdx.x >> 5;
  const int i    = blockIdx.x >> 4;
  const int jb   = blockIdx.x & 15;

  // ---------------- Phase 1 ----------------
  const int j0 = jb * 16 + wave * 2;        // this wave's two j's
  const __bf16* ap  = a_frag + ((size_t)i        * 16) * 512 + lane * 16;
  const __bf16* bp0 = b_frag + ((size_t)j0       * 16) * 512 + lane * 16;
  const __bf16* bp1 = b_frag + ((size_t)(j0 + 1) * 16) * 512 + lane * 16;

  v8f acc000 = {}, acc001 = {}, acc010 = {}, acc011 = {};   // j0: [ct][dt]
  v8f acc100 = {}, acc101 = {}, acc110 = {}, acc111 = {};   // j1: [ct][dt]

  #pragma unroll
  for (int kb = 0; kb < 8; ++kb) {
    const size_t o = (size_t)kb * 2 * 512;
    v16bf a0  = *(const v16bf*)(ap  + o);
    v16bf a1  = *(const v16bf*)(ap  + o + 512);
    v16bf b00 = *(const v16bf*)(bp0 + o);
    v16bf b01 = *(const v16bf*)(bp0 + o + 512);
    v16bf b10 = *(const v16bf*)(bp1 + o);
    v16bf b11 = *(const v16bf*)(bp1 + o + 512);
    acc000 = wmma_bf16(a0, b00, acc000);
    acc001 = wmma_bf16(a0, b01, acc001);
    acc010 = wmma_bf16(a1, b00, acc010);
    acc011 = wmma_bf16(a1, b01, acc011);
    acc100 = wmma_bf16(a0, b10, acc100);
    acc101 = wmma_bf16(a0, b11, acc101);
    acc110 = wmma_bf16(a1, b10, acc110);
    acc111 = wmma_bf16(a1, b11, acc111);
  }

  // Stash to LDS as bf16 in B-fragment order (converted exactly once).
  // C/D layout: vgpr r, lane L -> row (c) = ct*16 + r + 8*(L>=16), col (d) = dt*16 + (L&15)
  {
    const int rbase = 8 * (lane >> 4);
    const int dlow  = lane & 15;
    __bf16* l0 = &outer_lds[(wave * 2 + 0) * JSTRIDE_H + rbase * 32 + dlow];
    __bf16* l1 = &outer_lds[(wave * 2 + 1) * JSTRIDE_H + rbase * 32 + dlow];
    #pragma unroll
    for (int r = 0; r < 8; ++r) {
      l0[(r)      * 32]      = (__bf16)acc000[r];
      l0[(r)      * 32 + 16] = (__bf16)acc001[r];
      l0[(r + 16) * 32]      = (__bf16)acc010[r];
      l0[(r + 16) * 32 + 16] = (__bf16)acc011[r];
      l1[(r)      * 32]      = (__bf16)acc100[r];
      l1[(r)      * 32 + 16] = (__bf16)acc101[r];
      l1[(r + 16) * 32]      = (__bf16)acc110[r];
      l1[(r + 16) * 32 + 16] = (__bf16)acc111[r];
    }
  }

  __syncthreads();

  // ---------------- Phase 2 ----------------
  const int ztile = wave;                   // 8 waves x 16 rows = 128 z
  const int nn    = lane & 15;              // j_loc (B operand N index)
  const int khalf = (lane >> 4) << 4;       // K half within a 32-block
  const __bf16* wp = wfrag + ((size_t)ztile * 32) * 512 + lane * 16;
  const __bf16* lp = &outer_lds[nn * JSTRIDE_H + khalf];

  v8f acc2 = {};
  #pragma unroll 8
  for (int kb = 0; kb < 32; ++kb) {         // K = 1024: c = kb, d = khalf + e
    v16bf av = *(const v16bf*)(wp + (size_t)kb * 512);
    v16bf bv = *(const v16bf*)(lp + kb * 32);
    acc2 = wmma_bf16(av, bv, acc2);
  }

  // Coalesced store: each lane owns 8 consecutive z -> two float4 stores.
  {
    const int z0 = ztile * 16 + 8 * (lane >> 4);
    const int j  = jb * 16 + nn;
    const float sc = 1.0f / 256.0f;
    float4 bo0 = *(const float4*)(b_out + z0);
    float4 bo1 = *(const float4*)(b_out + z0 + 4);
    float4 r0  = { acc2[0] * sc + bo0.x, acc2[1] * sc + bo0.y,
                   acc2[2] * sc + bo0.z, acc2[3] * sc + bo0.w };
    float4 r1  = { acc2[4] * sc + bo1.x, acc2[5] * sc + bo1.y,
                   acc2[6] * sc + bo1.z, acc2[7] * sc + bo1.w };
    float* op = out + (((size_t)i * 256) + j) * 128 + z0;
    *(float4*)(op)     = r0;
    *(float4*)(op + 4) = r1;
  }
}

// ---------------------------------------------------------------------------
extern "C" void kernel_launch(void* const* d_in, const int* in_sizes, int n_in,
                              void* d_out, int out_size, void* d_ws, size_t ws_size,
                              hipStream_t stream) {
  (void)in_sizes; (void)n_in; (void)out_size; (void)ws_size;

  const float* m_si  = (const float*)d_in[0];
  const float* gamma = (const float*)d_in[1];
  const float* beta  = (const float*)d_in[2];
  const float* W_ab  = (const float*)d_in[3];
  const float* W_out = (const float*)d_in[4];
  const float* b_out = (const float*)d_in[5];
  float*       out   = (float*)d_out;

  char* ws = (char*)d_ws;
  __bf16* a_frag = (__bf16*)(ws);                    // 4 MB
  __bf16* b_frag = (__bf16*)(ws + (4u << 20));       // 4 MB
  __bf16* wfrag  = (__bf16*)(ws + (8u << 20));       // 256 KB

  ln_proj_kernel   <<<65536 / 8,    256, 0, stream>>>(m_si, gamma, beta, W_ab, a_frag, b_frag);
  wout_prep_kernel <<<131072 / 256, 256, 0, stream>>>(W_out, wfrag);
  outer_proj_kernel<<<4096,         256, 0, stream>>>(a_frag, b_frag, wfrag, b_out, out);
}
